// BatchEmbedding_32650341384803
// MI455X (gfx1250) — matrix operand order repaired and checked by reading, not compiled, not run on hardware
//
#include <hip/hip_runtime.h>
#include <hip/hip_bf16.h>

typedef _Float16 v16h __attribute__((ext_vector_type(16)));
typedef float    v8f  __attribute__((ext_vector_type(8)));

#define WAVES 8

// One workgroup (256 threads = 8 wave32) per batch row b.
// Tokens are counting-sorted by expert (deterministic), processed in
// 16-token single-expert tiles with v_wmma_f32_16x16x32_f16.
__global__ __launch_bounds__(256)
void BatchEmbedding_moe_kernel(const float* __restrict__ x,
                               const float* __restrict__ W1, const float* __restrict__ B1,
                               const float* __restrict__ W2, const float* __restrict__ B2,
                               const float* __restrict__ W3, const float* __restrict__ B3,
                               float* __restrict__ out)
{
    // ---------------- LDS ----------------
    __shared__ float    w1_lds[128];           // [E=4][32]
    __shared__ float    b1_lds[128];           // [4][32]
    __shared__ float    b2_lds[256];           // [4][64]
    __shared__ float    b3_lds[512];           // [4][128]
    __shared__ _Float16 w2h[8192];             // [4][n=64][k=32]  (transposed, f16)
    __shared__ _Float16 w3h[32768];            // [4][n=128][k=64] (transposed, f16)
    __shared__ float    v_lds[256];
    __shared__ float    R_lds[1024];           // [256][4]
    __shared__ int      sortedIdx[256];
    __shared__ _Float16 h2s[WAVES * 1024];     // per-wave 16x64 f16 staging
    __shared__ float    Pstage[WAVES * 512];   // per-wave partial P [128][4]
    __shared__ float    Pred[512];             // reduced P [128][4]
    __shared__ int      waveCnt[WAVES][4];
    __shared__ int      waveOff[WAVES][4];
    __shared__ int      tileE[20], tileS[20], tileC[20];
    __shared__ int      nTilesSh;

    const int  tid  = threadIdx.x;
    const int  wave = tid >> 5;
    const int  lane = tid & 31;
    const int  b    = blockIdx.x;
    const long base = (long)b * 1537;

    // ---- preload weights into LDS (transpose + f16 convert W2/W3) ----
    if (tid < 128) { w1_lds[tid] = W1[tid]; b1_lds[tid] = B1[tid]; }
    b2_lds[tid]       = B2[tid];
    b3_lds[tid]       = B3[tid];
    b3_lds[tid + 256] = B3[tid + 256];
    for (int idx = tid; idx < 8192; idx += 256) {       // w2h[e][n][k] = W2[e][k][n]
        int e = idx >> 11, r = idx & 2047, n = r >> 5, k = r & 31;
        w2h[idx] = (_Float16)W2[(e << 11) + k * 64 + n];
    }
    for (int idx = tid; idx < 32768; idx += 256) {      // w3h[e][n][k] = W3[e][k][n]
        int e = idx >> 13, r = idx & 8191, n = r >> 6, k = r & 63;
        w3h[idx] = (_Float16)W3[(e << 13) + k * 128 + n];
    }

    // ---- load token data for this row ----
    float vv = x[base + 2 * tid];
    float tv = x[base + 2 * tid + 1];
    v_lds[tid] = vv;
    #pragma unroll
    for (int c = 0; c < 4; ++c) R_lds[tid * 4 + c] = x[base + 512 + 4 * tid + c];
    int N = (int)x[base + 1536];
    if (N < 0) N = 0; if (N > 256) N = 256;
    int te = (int)tv; te = te < 0 ? 0 : (te > 3 ? 3 : te);
    const bool valid = tid < N;

    // ---- deterministic stable counting sort by expert ----
    int myPrefix = 0;
    #pragma unroll
    for (int e = 0; e < 4; ++e) {
        unsigned bal = (unsigned)__ballot(valid && (te == e));
        if (lane == 0) waveCnt[wave][e] = __popc(bal);
        if (te == e)   myPrefix = __popc(bal & ((1u << lane) - 1u));
    }
    __syncthreads();
    if (tid == 0) {
        int basePos = 0, nt = 0;
        for (int e = 0; e < 4; ++e) {
            int tot = 0;
            for (int w = 0; w < WAVES; ++w) { waveOff[w][e] = basePos + tot; tot += waveCnt[w][e]; }
            for (int s = 0; s < tot; s += 16) {
                tileE[nt] = e; tileS[nt] = basePos + s;
                tileC[nt] = (tot - s < 16) ? (tot - s) : 16; ++nt;
            }
            basePos += tot;
        }
        nTilesSh = nt;
    }
    __syncthreads();
    if (valid) sortedIdx[waveOff[wave][te] + myPrefix] = tid;
    __syncthreads();
    const int nTiles = nTilesSh;

    float pacc[32];                     // P partials: [ntile3 (8)][c (4)]
    #pragma unroll
    for (int i = 0; i < 32; ++i) pacc[i] = 0.f;

    const int mrow = lane & 15;         // matrix row / N-column owned by this lane
    const int half = lane >> 4;         // lane-group (K/M split per WMMA layouts)

    // ---- per-wave tile loop ----
    for (int ti = wave; ti < nTiles; ti += WAVES) {
        const int e = tileE[ti], ts = tileS[ti], tc = tileC[ti];

        // Layer 1 (scalar*vector) straight into 16x32 f16 A-fragment.
        float vm = (mrow < tc) ? v_lds[sortedIdx[ts + mrow]] : 0.f;
        v16h a1;
        #pragma unroll
        for (int i = 0; i < 16; ++i) {
            int k = half * 16 + i;
            float h = vm * w1_lds[e * 32 + k] + b1_lds[e * 32 + k];
            a1[i] = (_Float16)fmaxf(h, 0.f);
        }

        // Layer 2: (16x32) x (32x64) via 4 WMMAs; relu+bias; stage h2 f16.
        const int h2base = wave * 1024;
        #pragma unroll
        for (int nt2 = 0; nt2 < 4; ++nt2) {
            int n  = nt2 * 16 + mrow;
            int wb = e * 2048 + n * 32 + half * 16;
            v16h bf;
            #pragma unroll
            for (int i = 0; i < 16; ++i) bf[i] = w2h[wb + i];
            v8f c = {};
            c = __builtin_amdgcn_wmma_f32_16x16x32_f16(false, a1, false, bf,
                                                       (short)0, c, false, false);
            float bias = b2_lds[e * 64 + n];
            #pragma unroll
            for (int r = 0; r < 8; ++r) {
                int m = r + 8 * half;
                h2s[h2base + m * 64 + n] = (_Float16)fmaxf(c[r] + bias, 0.f);
            }
        }

        // A-fragments for layer 3 (16x64 split into two 16x32 halves).
        v16h a30, a31;
        #pragma unroll
        for (int i = 0; i < 16; ++i) {
            int k = half * 16 + i;
            a30[i] = h2s[h2base + mrow * 64 + k];
            a31[i] = h2s[h2base + mrow * 64 + 32 + k];
        }

        // R values for this tile's tokens (zero for padding -> kills contribution).
        float Rr[32];
        #pragma unroll
        for (int r = 0; r < 8; ++r) {
            int mr = r + 8 * half;
            if (mr < tc) {
                int tok = sortedIdx[ts + mr];
                #pragma unroll
                for (int c4 = 0; c4 < 4; ++c4) Rr[r * 4 + c4] = R_lds[tok * 4 + c4];
            } else {
                #pragma unroll
                for (int c4 = 0; c4 < 4; ++c4) Rr[r * 4 + c4] = 0.f;
            }
        }

        // Layer 3: (16x64) x (64x128) via 16 WMMAs, fused P accumulation.
        #pragma unroll
        for (int nt3 = 0; nt3 < 8; ++nt3) {
            int n  = nt3 * 16 + mrow;
            int wb = e * 8192 + n * 64 + half * 16;
            v16h b0, b1v;
            #pragma unroll
            for (int i = 0; i < 16; ++i) { b0[i] = w3h[wb + i]; b1v[i] = w3h[wb + 32 + i]; }
            v8f acc = {};
            acc = __builtin_amdgcn_wmma_f32_16x16x32_f16(false, a30, false, b0,
                                                         (short)0, acc, false, false);
            acc = __builtin_amdgcn_wmma_f32_16x16x32_f16(false, a31, false, b1v,
                                                         (short)0, acc, false, false);
            float bias = b3_lds[e * 128 + n];
            #pragma unroll
            for (int r = 0; r < 8; ++r) {
                float g = acc[r] + bias;
                #pragma unroll
                for (int c4 = 0; c4 < 4; ++c4)
                    pacc[nt3 * 4 + c4] = fmaf(g, Rr[r * 4 + c4], pacc[nt3 * 4 + c4]);
            }
        }
    }

    // ---- reduce P: half-wave, then cross-wave (fixed order -> deterministic) ----
    #pragma unroll
    for (int i = 0; i < 32; ++i) pacc[i] += __shfl_xor(pacc[i], 16, 32);
    if (half == 0) {
        #pragma unroll
        for (int nt = 0; nt < 8; ++nt)
            #pragma unroll
            for (int c4 = 0; c4 < 4; ++c4)
                Pstage[wave * 512 + (nt * 16 + mrow) * 4 + c4] = pacc[nt * 4 + c4];
    }
    __syncthreads();
    for (int idx = tid; idx < 512; idx += 256) {
        float s = 0.f;
        #pragma unroll
        for (int w = 0; w < WAVES; ++w) s += Pstage[w * 512 + idx];
        Pred[idx] = s;
    }
    __syncthreads();

    // ---- D[m1][m2] = sum_c P[m1][c] * P[m2][c]   (Q == P[:16,:]^T) ----
    const int m2 = tid & 15;
    const float q0 = Pred[m2 * 4 + 0], q1 = Pred[m2 * 4 + 1];
    const float q2 = Pred[m2 * 4 + 2], q3 = Pred[m2 * 4 + 3];
    #pragma unroll
    for (int j = 0; j < 8; ++j) {
        int m1 = (tid >> 4) + 16 * j;
        float d = Pred[m1 * 4 + 0] * q0 + Pred[m1 * 4 + 1] * q1
                + Pred[m1 * 4 + 2] * q2 + Pred[m1 * 4 + 3] * q3;
        out[(long)b * 2048 + m1 * 16 + m2] = d;
    }
}

extern "C" void kernel_launch(void* const* d_in, const int* in_sizes, int n_in,
                              void* d_out, int out_size, void* d_ws, size_t ws_size,
                              hipStream_t stream) {
    const float* x  = (const float*)d_in[0];
    const float* W1 = (const float*)d_in[1];
    const float* B1 = (const float*)d_in[2];
    const float* W2 = (const float*)d_in[3];
    const float* B2 = (const float*)d_in[4];
    const float* W3 = (const float*)d_in[5];
    const float* B3 = (const float*)d_in[6];
    float* out = (float*)d_out;

    const int Bsz = in_sizes[0] / 1537;   // x is (B, 6*A+1), A=256
    BatchEmbedding_moe_kernel<<<dim3(Bsz), dim3(256), 0, stream>>>(
        x, W1, B1, W2, B2, W3, B3, out);
}
